// LinearAttention_87840671138277
// MI455X (gfx1250) — matrix operand order repaired and checked by reading, not compiled, not run on hardware
//
#include <hip/hip_runtime.h>

#define BB 8
#define NN 16384
#define CC 256
#define DD 32
#define EPSF 1e-6f

typedef __attribute__((ext_vector_type(16))) __bf16 v16bf;
typedef __attribute__((ext_vector_type(8)))  float  v8f;

union Frag {
  unsigned u[8];
  uint4    q[2];
  v16bf    b;
};

__device__ inline v8f v8zero() {
  v8f z;
#pragma unroll
  for (int i = 0; i < 8; ++i) z[i] = 0.f;
  return z;
}

// fp32 -> bf16 (round to nearest even) helpers
__device__ inline unsigned short f2bf(float f) {
  unsigned u = __float_as_uint(f);
  u += 0x7FFFu + ((u >> 16) & 1u);
  return (unsigned short)(u >> 16);
}

#if defined(__has_builtin)
#if __has_builtin(__builtin_amdgcn_cvt_pk_bf16_f32)
#define HAVE_CVT_PK_BF16 1
#endif
#endif

__device__ inline unsigned pk2(float lo, float hi) {
#ifdef HAVE_CVT_PK_BF16
  return __builtin_bit_cast(unsigned, __builtin_amdgcn_cvt_pk_bf16_f32(lo, hi));
#else
  return (unsigned)f2bf(lo) | ((unsigned)f2bf(hi) << 16);
#endif
}
__device__ inline float bflo(unsigned u) { return __uint_as_float(u << 16); }
__device__ inline float bfhi(unsigned u) { return __uint_as_float(u & 0xFFFF0000u); }

__device__ inline v8f wmma_bf16(const Frag& a, const Frag& bm, v8f c) {
  return __builtin_amdgcn_wmma_f32_16x16x32_bf16(false, a.b, false, bm.b,
                                                 (short)0, c, false, false);
}

// ---------------------------------------------------------------------------
// Kernel 1: fused Q/K/V projection (bf16 WMMA), per-position L2 norm of Q,K,
// Q stored bf16; Ksum, vsum and mat = K_norm * V^T reduced over n via WMMA
// (ones-row trick gives vsum as row 32 of mat). V bias deliberately omitted
// (restored analytically in kernel 2).
// LDS: sQK 32KB | sWv 128KB | sX 16KB | sKst 2KB | sVt 8KB  = 190464 B
// ---------------------------------------------------------------------------
__global__ __launch_bounds__(320) void la_proj(
    const float* __restrict__ x, const float* __restrict__ y,
    const float* __restrict__ Wq, const float* __restrict__ bq,
    const float* __restrict__ Wk, const float* __restrict__ bk,
    const float* __restrict__ Wv,
    const float* __restrict__ Wqy, const float* __restrict__ bqy,
    const float* __restrict__ Wky, const float* __restrict__ bky,
    const float* __restrict__ Wvy,
    unsigned short* __restrict__ Qws, float* __restrict__ matWs,
    float* __restrict__ ksumWs)
{
  extern __shared__ char smem[];
  unsigned short* sQK  = (unsigned short*)(smem);            // [64][256] Wq|Wk
  unsigned short* sWv  = (unsigned short*)(smem + 32768);    // [256][256]
  unsigned short* sX   = (unsigned short*)(smem + 163840);   // [32][256]
  unsigned short* sKst = (unsigned short*)(smem + 180224);   // [32][32] K-norm
  unsigned short* sVt  = (unsigned short*)(smem + 182272);   // [8][16][32]

  const int tid  = threadIdx.x;
  const int wid  = tid >> 5;
  const int lane = tid & 31;
  const int l15  = lane & 15;
  const int lh   = (lane >> 4) & 1;
  const int s = blockIdx.y;
  const int b = blockIdx.z;
  const int chunk = blockIdx.x;

  const float* xb  = (s ? y : x) + (size_t)b * NN * CC;
  const float* Wq_ = s ? Wqy : Wq;  const float* bq_ = s ? bqy : bq;
  const float* Wk_ = s ? Wky : Wk;  const float* bk_ = s ? bky : bk;
  const float* Wv_ = s ? Wvy : Wv;

  // stage weights (fp32 -> bf16) once
  for (int i = tid; i < 64 * 64; i += 320) {
    int r = i >> 6, c4 = (i & 63) << 2;
    const float* src = (r < 32) ? (Wq_ + r * CC + c4) : (Wk_ + (r - 32) * CC + c4);
    float4 v = *(const float4*)src;
    *(uint2*)(sQK + r * 256 + c4) = make_uint2(pk2(v.x, v.y), pk2(v.z, v.w));
  }
  for (int i = tid; i < 256 * 64; i += 320) {
    int r = i >> 6, c4 = (i & 63) << 2;
    float4 v = *(const float4*)(Wv_ + (size_t)r * CC + c4);
    *(uint2*)(sWv + r * 256 + c4) = make_uint2(pk2(v.x, v.y), pk2(v.z, v.w));
  }

  // persistent per-wave state
  float bqr[16], bkr[16], ksumAcc[16];
  v8f mA0[2], mA1[2], mAS[2];
  Frag aOne;
  if (wid < 2) {
#pragma unroll
    for (int t = 0; t < 2; ++t)
#pragma unroll
      for (int v = 0; v < 8; ++v) {
        int d = t * 16 + lh * 8 + v;
        bqr[t * 8 + v] = bq_[d];
        bkr[t * 8 + v] = bk_[d];
        ksumAcc[t * 8 + v] = 0.f;
      }
  } else {
#pragma unroll
    for (int i = 0; i < 2; ++i) { mA0[i] = v8zero(); mA1[i] = v8zero(); mAS[i] = v8zero(); }
#pragma unroll
    for (int v = 0; v < 8; ++v) aOne.u[v] = (l15 == 0) ? 0x3F803F80u : 0u;  // row M=0 = 1.0
  }
  __syncthreads();

  const int NSTEPS = 32;
  for (int stp = 0; stp < NSTEPS; ++stp) {
    const int n0 = chunk * 1024 + stp * 32;
    // cooperative stage of x[n0..n0+31][0..255] as bf16
    for (int i = tid; i < 32 * 64; i += 320) {
      int r = i >> 6, c4 = (i & 63) << 2;
      float4 v = *(const float4*)(xb + (size_t)(n0 + r) * CC + c4);
      *(uint2*)(sX + r * 256 + c4) = make_uint2(pk2(v.x, v.y), pk2(v.z, v.w));
    }
    // prefetch next 32x256 fp32 tile (one 128B line per thread)
    if (stp + 1 < NSTEPS && tid < 256)
      __builtin_prefetch(xb + (size_t)(n0 + 32) * CC + tid * 32, 0, 1);
    __syncthreads();

    Frag bV[2];  // V-wave mat B fragments (live across the mid barrier)
    if (wid < 2) {
      // ---- Q/K projection for 16 positions (half h = wid) ----
      v8f cQ0, cQ1, cK0, cK1;
#pragma unroll
      for (int v = 0; v < 8; ++v) {
        cQ0[v] = bqr[v]; cQ1[v] = bqr[8 + v];
        cK0[v] = bkr[v]; cK1[v] = bkr[8 + v];
      }
      const int xrow = wid * 16 + l15;
#pragma unroll
      for (int kc = 0; kc < 8; ++kc) {
        Frag bX, aT;
        const unsigned short* bp = sX + xrow * 256 + kc * 32 + lh * 8;
        bX.q[0] = *(const uint4*)bp;  bX.q[1] = *(const uint4*)(bp + 16);
        const int aoff = kc * 32 + lh * 8;
        const unsigned short* ap;
        ap = sQK + (l15)      * 256 + aoff; aT.q[0] = *(const uint4*)ap; aT.q[1] = *(const uint4*)(ap + 16);
        cQ0 = wmma_bf16(aT, bX, cQ0);
        ap = sQK + (16 + l15) * 256 + aoff; aT.q[0] = *(const uint4*)ap; aT.q[1] = *(const uint4*)(ap + 16);
        cQ1 = wmma_bf16(aT, bX, cQ1);
        ap = sQK + (32 + l15) * 256 + aoff; aT.q[0] = *(const uint4*)ap; aT.q[1] = *(const uint4*)(ap + 16);
        cK0 = wmma_bf16(aT, bX, cK0);
        ap = sQK + (48 + l15) * 256 + aoff; aT.q[0] = *(const uint4*)ap; aT.q[1] = *(const uint4*)(ap + 16);
        cK1 = wmma_bf16(aT, bX, cK1);
      }
      // per-position L2 norm over d=32: lane pair (l, l+16) holds the column
      float qs = 0.f, ks = 0.f;
#pragma unroll
      for (int v = 0; v < 8; ++v) {
        qs += cQ0[v] * cQ0[v] + cQ1[v] * cQ1[v];
        ks += cK0[v] * cK0[v] + cK1[v] * cK1[v];
      }
      qs += __shfl_xor(qs, 16, 32);
      ks += __shfl_xor(ks, 16, 32);
      const float rq = rsqrtf(qs), rk = rsqrtf(ks);
      const int n = n0 + wid * 16 + l15;
      unsigned short* qrow = Qws + ((size_t)(s * BB + b) * NN + n) * DD;
      uint4 o0 = make_uint4(pk2(cQ0[0]*rq, cQ0[1]*rq), pk2(cQ0[2]*rq, cQ0[3]*rq),
                            pk2(cQ0[4]*rq, cQ0[5]*rq), pk2(cQ0[6]*rq, cQ0[7]*rq));
      *(uint4*)(qrow + lh * 8) = o0;
      uint4 o1 = make_uint4(pk2(cQ1[0]*rq, cQ1[1]*rq), pk2(cQ1[2]*rq, cQ1[3]*rq),
                            pk2(cQ1[4]*rq, cQ1[5]*rq), pk2(cQ1[6]*rq, cQ1[7]*rq));
      *(uint4*)(qrow + 16 + lh * 8) = o1;
      const int ncol = wid * 16 + l15;
#pragma unroll
      for (int v = 0; v < 8; ++v) {
        float k0 = cK0[v] * rk, k1 = cK1[v] * rk;
        ksumAcc[v] += k0; ksumAcc[8 + v] += k1;
        sKst[(lh * 8 + v) * 32 + ncol]        = f2bf(k0);
        sKst[(16 + lh * 8 + v) * 32 + ncol]   = f2bf(k1);
      }
    } else {
      // ---- V projection for this wave's 2 c-tiles, build mat-B fragments ----
      const int vw = wid - 2;
      unsigned short* myVt = sVt + vw * 512;  // [16][32]
#pragma unroll
      for (int ctl = 0; ctl < 2; ++ctl) {
        const int ct = vw * 2 + ctl;
#pragma unroll
        for (int h = 0; h < 2; ++h) {
          v8f cV = v8zero();
#pragma unroll
          for (int kc = 0; kc < 8; ++kc) {
            Frag bX, aW;
            const unsigned short* bp = sX + (h * 16 + l15) * 256 + kc * 32 + lh * 8;
            bX.q[0] = *(const uint4*)bp;  bX.q[1] = *(const uint4*)(bp + 16);
            const unsigned short* ap = sWv + (ct * 16 + l15) * 256 + kc * 32 + lh * 8;
            aW.q[0] = *(const uint4*)ap;  aW.q[1] = *(const uint4*)(ap + 16);
            cV = wmma_bf16(aW, bX, cV);
          }
#pragma unroll
          for (int v = 0; v < 8; ++v)
            myVt[(lh * 8 + v) * 32 + h * 16 + l15] = f2bf(cV[v]);
        }
        const unsigned short* rp = myVt + l15 * 32 + lh * 8;
        bV[ctl].q[0] = *(const uint4*)rp;  bV[ctl].q[1] = *(const uint4*)(rp + 16);
      }
    }
    __syncthreads();  // sKst ready

    if (wid >= 2) {
      Frag aK0, aK1;
      const unsigned short* kp0 = sKst + l15 * 32 + lh * 8;
      aK0.q[0] = *(const uint4*)kp0;        aK0.q[1] = *(const uint4*)(kp0 + 16);
      const unsigned short* kp1 = sKst + (16 + l15) * 32 + lh * 8;
      aK1.q[0] = *(const uint4*)kp1;        aK1.q[1] = *(const uint4*)(kp1 + 16);
#pragma unroll
      for (int ctl = 0; ctl < 2; ++ctl) {
        mA0[ctl] = wmma_bf16(aK0,  bV[ctl], mA0[ctl]);
        mA1[ctl] = wmma_bf16(aK1,  bV[ctl], mA1[ctl]);
        mAS[ctl] = wmma_bf16(aOne, bV[ctl], mAS[ctl]);   // vsum row
      }
    }
    __syncthreads();  // protect sX/sKst before next step
  }

  // ---- final reductions -> workspace atomics ----
  if (wid < 2) {
#pragma unroll
    for (int i = 0; i < 16; ++i) {
      float v = ksumAcc[i];
      v += __shfl_xor(v, 1, 32);  v += __shfl_xor(v, 2, 32);
      v += __shfl_xor(v, 4, 32);  v += __shfl_xor(v, 8, 32);
      ksumAcc[i] = v;
    }
    if (l15 == 0) {
      float* kp = ksumWs + (s * BB + b) * DD;
#pragma unroll
      for (int t = 0; t < 2; ++t)
#pragma unroll
        for (int v = 0; v < 8; ++v)
          atomicAdd(kp + t * 16 + lh * 8 + v, ksumAcc[t * 8 + v]);
    }
  } else {
    const int vw = wid - 2;
    float* mp = matWs + (size_t)(s * BB + b) * 33 * CC;
#pragma unroll
    for (int ctl = 0; ctl < 2; ++ctl) {
      const int c = (vw * 2 + ctl) * 16 + l15;
#pragma unroll
      for (int v = 0; v < 8; ++v) {
        atomicAdd(mp + (size_t)(lh * 8 + v) * CC + c,      mA0[ctl][v]);
        atomicAdd(mp + (size_t)(16 + lh * 8 + v) * CC + c, mA1[ctl][v]);
      }
      if (lh == 0) atomicAdd(mp + (size_t)32 * CC + c, mAS[ctl][0]);
    }
  }
}

// ---------------------------------------------------------------------------
// Kernel 2: out[n,c] for one stream s:
//   fx = t1 * (vsum_s + Q.mat_s) + t2 * (vsum_s + Q.mat_{1-s}),
//   t_i = gamma_i*w_i / (N + Q.(Ksum_i + eps)).
// Mats staged transposed+bias-corrected+vsum-augmented in LDS ([c][64] bf16,
// row 32 = vsum, rows 33..63 = 0); Q read as B-fragment straight from global.
// ---------------------------------------------------------------------------
__global__ __launch_bounds__(256) void la_out(
    const unsigned short* __restrict__ Qws, const float* __restrict__ matWs,
    const float* __restrict__ ksumWs,
    const float* __restrict__ bv, const float* __restrict__ bvy,
    const float* __restrict__ gamma, const float* __restrict__ gamma_y,
    const float* __restrict__ gamma_cx, const float* __restrict__ gamma_cy,
    const float* __restrict__ wx1, const float* __restrict__ wx2,
    const float* __restrict__ wy1, const float* __restrict__ wy2,
    float* __restrict__ out)
{
  extern __shared__ char smem[];
  unsigned short* sMat = (unsigned short*)smem;      // [2][256][64]
  float* sKs1 = (float*)(smem + 65536);
  float* sKs2 = sKs1 + 32;

  const int tid  = threadIdx.x;
  const int wid  = tid >> 5;
  const int lane = tid & 31;
  const int l15  = lane & 15;
  const int lh   = (lane >> 4) & 1;
  const int s = blockIdx.y, b = blockIdx.z, chunk = blockIdx.x;

  const float* matP = matWs + (size_t)(s * BB + b) * 33 * CC;
  const float* matS = matWs + (size_t)((1 - s) * BB + b) * 33 * CC;
  const float* ksP = ksumWs + (s * BB + b) * DD;
  const float* ksS = ksumWs + ((1 - s) * BB + b) * DD;
  const float* bvP = s ? bvy : bv;
  const float* bvS = s ? bv : bvy;

  for (int i = tid; i < 2 * 64 * 256; i += 256) {
    const int m = i >> 14, r = (i >> 8) & 63, c = i & 255;
    float val = 0.f;
    if (r < 32) {
      if (m == 0) val = matP[r * CC + c] + bvP[c] * ksP[r];   // restore V bias
      else        val = matS[r * CC + c] + bvS[c] * ksS[r];
    } else if (r == 32) {
      val = matP[32 * CC + c] + (float)NN * bvP[c];           // vsum of stream s
    }
    sMat[m * 16384 + c * 64 + r] = f2bf(val);
  }
  if (tid < 64) {
    const int h = tid >> 5, d = tid & 31;
    const float v = (h ? ksS[d] : ksP[d]) + EPSF;
    (h ? sKs2 : sKs1)[d] = v;
  }
  __syncthreads();

  float g1, g2;
  if (s == 0) { g1 = gamma[0]   * wx1[0]; g2 = gamma_cx[0] * wx2[0]; }
  else        { g1 = gamma_y[0] * wy1[0]; g2 = gamma_cy[0] * wy2[0]; }

  Frag b2;  // ones column at d=32 (vsum augmentation)
#pragma unroll
  for (int v = 0; v < 8; ++v) b2.u[v] = 0u;
  if (lh == 0) b2.u[0] = 0x00003F80u;

  float* obase = out + ((size_t)s * BB + b) * NN * CC;

  for (int stp = 0; stp < 8; ++stp) {
    const int n = chunk * 1024 + (stp * 8 + wid) * 16 + l15;
    const unsigned short* qp =
        Qws + ((size_t)(s * BB + b) * NN + n) * DD + lh * 16;
    Frag bQ;
    bQ.q[0] = *(const uint4*)qp;  bQ.q[1] = *(const uint4*)(qp + 8);
    float d1 = 0.f, d2 = 0.f;
#pragma unroll
    for (int v = 0; v < 8; ++v) {
      const unsigned u = bQ.u[v];
      const float flo = bflo(u), fhi = bfhi(u);
      const int dd = lh * 16 + 2 * v;
      d1 += flo * sKs1[dd] + fhi * sKs1[dd + 1];
      d2 += flo * sKs2[dd] + fhi * sKs2[dd + 1];
    }
    d1 += __shfl_xor(d1, 16, 32);
    d2 += __shfl_xor(d2, 16, 32);
    const float t1 = g1 / ((float)NN + d1);
    const float t2 = g2 / ((float)NN + d2);
    float* orow = obase + (size_t)n * CC;

#pragma unroll 4
    for (int ct = 0; ct < 16; ++ct) {
      const unsigned short* a1p = sMat + (ct * 16 + l15) * 64 + lh * 16;
      const unsigned short* a2p = a1p + 16384;
      Frag A;
      v8f C1 = v8zero(), C2 = v8zero();
      A.q[0] = *(const uint4*)a1p;        A.q[1] = *(const uint4*)(a1p + 8);
      C1 = wmma_bf16(A, bQ, C1);
      A.q[0] = *(const uint4*)(a1p + 32); A.q[1] = *(const uint4*)(a1p + 40);
      C1 = wmma_bf16(A, b2, C1);
      A.q[0] = *(const uint4*)a2p;        A.q[1] = *(const uint4*)(a2p + 8);
      C2 = wmma_bf16(A, bQ, C2);
      A.q[0] = *(const uint4*)(a2p + 32); A.q[1] = *(const uint4*)(a2p + 40);
      C2 = wmma_bf16(A, b2, C2);
      float4 r0, r1;
      r0.x = t1 * C1[0] + t2 * C2[0];  r0.y = t1 * C1[1] + t2 * C2[1];
      r0.z = t1 * C1[2] + t2 * C2[2];  r0.w = t1 * C1[3] + t2 * C2[3];
      r1.x = t1 * C1[4] + t2 * C2[4];  r1.y = t1 * C1[5] + t2 * C2[5];
      r1.z = t1 * C1[6] + t2 * C2[6];  r1.w = t1 * C1[7] + t2 * C2[7];
      *(float4*)(orow + ct * 16 + lh * 8)     = r0;
      *(float4*)(orow + ct * 16 + lh * 8 + 4) = r1;
    }
  }
}

extern "C" void kernel_launch(void* const* d_in, const int* in_sizes, int n_in,
                              void* d_out, int out_size, void* d_ws, size_t ws_size,
                              hipStream_t stream) {
  (void)in_sizes; (void)n_in; (void)out_size; (void)ws_size;
  const float* x   = (const float*)d_in[0];
  const float* y   = (const float*)d_in[1];
  const float* Wq  = (const float*)d_in[2];  const float* bq  = (const float*)d_in[3];
  const float* Wk  = (const float*)d_in[4];  const float* bk  = (const float*)d_in[5];
  const float* Wv  = (const float*)d_in[6];  const float* bv  = (const float*)d_in[7];
  const float* Wqy = (const float*)d_in[8];  const float* bqy = (const float*)d_in[9];
  const float* Wky = (const float*)d_in[10]; const float* bky = (const float*)d_in[11];
  const float* Wvy = (const float*)d_in[12]; const float* bvy = (const float*)d_in[13];
  const float* gamma    = (const float*)d_in[14];
  const float* gamma_y  = (const float*)d_in[15];
  const float* gamma_cx = (const float*)d_in[16];
  const float* gamma_cy = (const float*)d_in[17];
  const float* wx1 = (const float*)d_in[18];
  const float* wx2 = (const float*)d_in[19];
  const float* wy1 = (const float*)d_in[20];
  const float* wy2 = (const float*)d_in[21];
  float* out = (float*)d_out;

  char* ws = (char*)d_ws;
  const size_t QBYTES   = (size_t)2 * BB * NN * DD * 2;   // Q / Qy bf16
  const size_t MATBYTES = (size_t)2 * BB * 33 * CC * 4;   // mat + vsum row
  const size_t KSBYTES  = (size_t)2 * BB * DD * 4;        // Ksum / Kysum
  unsigned short* Qws = (unsigned short*)ws;
  float* matWs  = (float*)(ws + QBYTES);
  float* ksumWs = (float*)(ws + QBYTES + MATBYTES);

  hipMemsetAsync(ws + QBYTES, 0, MATBYTES + KSBYTES, stream);

  (void)hipFuncSetAttribute((const void*)la_proj,
                            hipFuncAttributeMaxDynamicSharedMemorySize, 190464);
  (void)hipFuncSetAttribute((const void*)la_out,
                            hipFuncAttributeMaxDynamicSharedMemorySize, 65792);

  la_proj<<<dim3(16, 2, 8), 320, 190464, stream>>>(
      x, y, Wq, bq, Wk, bk, Wv, Wqy, bqy, Wky, bky, Wvy, Qws, matWs, ksumWs);
  la_out<<<dim3(16, 2, 8), 256, 65792, stream>>>(
      Qws, matWs, ksumWs, bv, bvy, gamma, gamma_y, gamma_cx, gamma_cy,
      wx1, wx2, wy1, wy2, out);
}
